// NanoGPTLeanBlock_6313601925682
// MI455X (gfx1250) — matrix operand order repaired and checked by reading, compiled, not run on hardware
//
#include <hip/hip_runtime.h>

// ---------------- problem constants ----------------
#define NB 2
#define NT 2048
#define ND 2048
#define NH 16
#define DH 128
#define NF 8192
#define NR (NB*NT)
#define EPSF 1.1920928955078125e-7f   // np.float32 eps

typedef __attribute__((ext_vector_type(16))) __bf16 v16bf;
typedef __attribute__((ext_vector_type(8)))  __bf16 v8bf;
typedef __attribute__((ext_vector_type(8)))  float  v8f;

static __device__ __forceinline__ v16bf cat16(v8bf lo, v8bf hi) {
  return __builtin_shufflevector(lo, hi, 0,1,2,3,4,5,6,7,8,9,10,11,12,13,14,15);
}
static __device__ __forceinline__ v8f wmma_bf16(v16bf a, v16bf b, v8f c) {
  // D = A(16x32) * B(32x16) + C, fp32 accumulate
  return __builtin_amdgcn_wmma_f32_16x16x32_bf16(false, a, false, b, (short)0, c, false, false);
}
static __device__ __forceinline__ v8f vzero8() {
  v8f z;
  #pragma unroll
  for (int i = 0; i < 8; ++i) z[i] = 0.0f;
  return z;
}
static __device__ __forceinline__ v8bf pack8(float4 a, float4 b) {
  v8bf p;
  p[0]=(__bf16)a.x; p[1]=(__bf16)a.y; p[2]=(__bf16)a.z; p[3]=(__bf16)a.w;
  p[4]=(__bf16)b.x; p[5]=(__bf16)b.y; p[6]=(__bf16)b.z; p[7]=(__bf16)b.w;
  return p;
}

// ---------------- RMSNorm (fp32 in -> bf16 out) ----------------
__global__ __launch_bounds__(256) void rmsnorm_to_bf16(
    const float* __restrict__ x, const float* __restrict__ w,
    __bf16* __restrict__ out, int ncols)
{
  const int row = blockIdx.x;
  const float* xr = x + (long)row * ncols;
  float ss = 0.f;
  for (int c = threadIdx.x; c < ncols; c += 256) { float v = xr[c]; ss += v * v; }
  #pragma unroll
  for (int off = 16; off > 0; off >>= 1) ss += __shfl_xor(ss, off, 32);
  __shared__ float red[8];
  if ((threadIdx.x & 31) == 0) red[threadIdx.x >> 5] = ss;
  __syncthreads();
  if (threadIdx.x == 0) {
    float tot = 0.f;
    #pragma unroll
    for (int i = 0; i < 8; ++i) tot += red[i];
    red[0] = rsqrtf(tot / (float)ncols + EPSF);
  }
  __syncthreads();
  const float inv = red[0];
  __bf16* orow = out + (long)row * ncols;
  for (int c = threadIdx.x; c < ncols; c += 256)
    orow[c] = (__bf16)(xr[c] * inv * w[c]);
}

// ---------------- GEMM: C[M,N] = act(A[MxK](bf16) @ W[NxK](f32)^T) (+resid) ----------------
// Tile 128x128x32, 8 waves, each wave computes 16(M) x 128(N) via 8 WMMA accs.
// Double-buffered LDS: global loads for tile k+1 are issued before the WMMA
// chain for tile k, staged to the alternate buffer after compute; one barrier
// per K-step. Weights converted f32->bf16 while staging (single weight pass).
__global__ __launch_bounds__(256) void gemm_bf16_wmma(
    const __bf16* __restrict__ A, const float* __restrict__ W,
    const float* __restrict__ resid, void* __restrict__ Cout,
    int M, int N, int K, int out_bf16, int act_sqrelu)
{
  __shared__ __align__(16) __bf16 As[2][128][32];
  __shared__ __align__(16) __bf16 Bs[2][128][32];
  const int tix  = threadIdx.x;
  const int wave = tix >> 5, lane = tix & 31;
  const int l15  = lane & 15;
  const int half = (lane >> 4) & 1;
  const long tM = (long)blockIdx.y * 128;
  const long tN = (long)blockIdx.x * 128;

  v8f acc[8];
  #pragma unroll
  for (int i = 0; i < 8; ++i) acc[i] = vzero8();

  const int ldr = tix >> 1;          // staging row 0..127
  const int ldh = tix & 1;           // which 16-element half
  const __bf16* Aptr = A + (tM + ldr) * (long)K + ldh * 16;
  const float*  Wptr = W + (tN + ldr) * (long)K + ldh * 16;

  // ---- prologue: fetch + stage tile 0
  uint4  a0 = *(const uint4*)(Aptr);
  uint4  a1 = *(const uint4*)(Aptr + 8);
  float4 w0 = *(const float4*)(Wptr);
  float4 w1 = *(const float4*)(Wptr + 4);
  float4 w2 = *(const float4*)(Wptr + 8);
  float4 w3 = *(const float4*)(Wptr + 12);
  int buf = 0;
  *(uint4*)&As[buf][ldr][ldh * 16]     = a0;
  *(uint4*)&As[buf][ldr][ldh * 16 + 8] = a1;
  *(v8bf*)&Bs[buf][ldr][ldh * 16]      = pack8(w0, w1);
  *(v8bf*)&Bs[buf][ldr][ldh * 16 + 8]  = pack8(w2, w3);
  __syncthreads();

  const int rowA = (wave << 4) + l15;
  for (int k0 = 0; k0 < K; k0 += 32) {
    const bool more = (k0 + 32) < K;
    if (more) {
      // issue next tile's global loads early (hidden under the WMMA chain)
      a0 = *(const uint4*)(Aptr + k0 + 32);
      a1 = *(const uint4*)(Aptr + k0 + 40);
      w0 = *(const float4*)(Wptr + k0 + 32);
      w1 = *(const float4*)(Wptr + k0 + 36);
      w2 = *(const float4*)(Wptr + k0 + 40);
      w3 = *(const float4*)(Wptr + k0 + 44);
      if (k0 + 64 < K) {
        __builtin_prefetch(Wptr + k0 + 64, 0, 1);   // global_prefetch_b8
        __builtin_prefetch(Aptr + k0 + 64, 0, 1);
      }
    }
    // ---- compute from LDS[buf]: preload all fragments, then WMMA chain
    v8bf alo = *(const v8bf*)&As[buf][rowA][half * 8];
    v8bf ahi = *(const v8bf*)&As[buf][rowA][16 + half * 8];
    v16bf af = cat16(alo, ahi);
    v16bf bfr[8];
    #pragma unroll
    for (int nt = 0; nt < 8; ++nt) {
      const int colB = (nt << 4) + l15;
      v8bf blo = *(const v8bf*)&Bs[buf][colB][half * 8];
      v8bf bhi = *(const v8bf*)&Bs[buf][colB][16 + half * 8];
      bfr[nt] = cat16(blo, bhi);
    }
    #pragma unroll
    for (int nt = 0; nt < 8; ++nt) acc[nt] = wmma_bf16(af, bfr[nt], acc[nt]);

    if (more) {
      buf ^= 1;
      *(uint4*)&As[buf][ldr][ldh * 16]     = a0;
      *(uint4*)&As[buf][ldr][ldh * 16 + 8] = a1;
      *(v8bf*)&Bs[buf][ldr][ldh * 16]      = pack8(w0, w1);
      *(v8bf*)&Bs[buf][ldr][ldh * 16 + 8]  = pack8(w2, w3);
      __syncthreads();
    }
  }

  // Epilogue per ISA C layout: lane holds col l15, rows half*8 + v
  const long rbase = tM + (wave << 4) + (half << 3);
  #pragma unroll
  for (int nt = 0; nt < 8; ++nt) {
    const long col = tN + (nt << 4) + l15;
    #pragma unroll
    for (int v = 0; v < 8; ++v) {
      const long row = rbase + v;
      float val = acc[nt][v];
      if (act_sqrelu) { val = fmaxf(val, 0.0f); val = val * val; }
      if (resid) val += resid[row * (long)N + col];
      if (out_bf16) ((__bf16*)Cout)[row * (long)N + col] = (__bf16)val;
      else          ((float*)Cout)[row * (long)N + col]  = val;
    }
  }
}

// ---------------- RoPE + Q/K RMSNorm + V transpose, fp32 -> bf16 ----------------
__global__ __launch_bounds__(128) void rope_qknorm(
    const float* __restrict__ qkv, const float* __restrict__ rope,
    const float* __restrict__ qw, const float* __restrict__ kw,
    __bf16* __restrict__ Qb, __bf16* __restrict__ Kb, __bf16* __restrict__ Vt)
{
  const int r = blockIdx.x;            // flattened (b,t)
  const int b = r / NT, t = r % NT;
  const int h = blockIdx.y;
  const int d = threadIdx.x;           // 0..127
  const long base = (long)r * (3 * ND) + h * DH + d;
  float q = qkv[base];
  float k = qkv[base + ND];
  float v = qkv[base + 2 * ND];
  const float c = rope[(long)t * DH + d];
  const float s = rope[(long)(NT + t) * DH + d];
  const float qp = __shfl_xor(q, 1, 32);   // pair partner (d ^ 1)
  const float kp = __shfl_xor(k, 1, 32);
  const float sgn = (d & 1) ? 1.0f : -1.0f;
  const float qr = q * c + sgn * qp * s;
  const float kr = k * c + sgn * kp * s;
  __shared__ float red[2][4];
  float sq = qr * qr, sk = kr * kr;
  #pragma unroll
  for (int off = 16; off > 0; off >>= 1) {
    sq += __shfl_xor(sq, off, 32);
    sk += __shfl_xor(sk, off, 32);
  }
  if ((d & 31) == 0) { red[0][d >> 5] = sq; red[1][d >> 5] = sk; }
  __syncthreads();
  sq = red[0][0] + red[0][1] + red[0][2] + red[0][3];
  sk = red[1][0] + red[1][1] + red[1][2] + red[1][3];
  const float qscale = 0.08838834764831843f;   // 1/sqrt(DH) folded into Q
  const float qn = qr * rsqrtf(sq * (1.0f / DH) + EPSF) * qw[d] * qscale;
  const float kn = kr * rsqrtf(sk * (1.0f / DH) + EPSF) * kw[d];
  const long bh = (long)b * NH + h;
  Qb[(bh * NT + t) * DH + d] = (__bf16)qn;
  Kb[(bh * NT + t) * DH + d] = (__bf16)kn;
  Vt[(bh * DH + d) * NT + t] = (__bf16)v;      // V stored [Dh, T] for PV B-frags
}

// ---------------- Flash attention: WMMA QK^T + online softmax + WMMA PV ----------------
// Block = 4 waves; each wave owns 16 query rows of a 64-row tile; 32 keys/step.
__global__ __launch_bounds__(128, 1) void flash_attn_wmma(
    const __bf16* __restrict__ Qb, const __bf16* __restrict__ Kb,
    const __bf16* __restrict__ Vt, __bf16* __restrict__ attn_out)
{
  __shared__ __align__(16) __bf16 Pl[4][16][32];
  const int wave = threadIdx.x >> 5, lane = threadIdx.x & 31;
  const int l15 = lane & 15, half = (lane >> 4) & 1;
  const int bh = blockIdx.y, b = bh >> 4, h = bh & 15;
  const int q0 = blockIdx.x * 64;
  const int qwb = q0 + wave * 16;

  const __bf16* Qp = Qb + (long)bh * NT * DH;
  const __bf16* Kp = Kb + (long)bh * NT * DH;
  const __bf16* Vp = Vt + (long)bh * DH * NT;

  // Q fragments (16 rows x 128 dh = 4 k-chunks), A layout
  v16bf qf[4];
  {
    const __bf16* qrow = Qp + (long)(qwb + l15) * DH + half * 8;
    #pragma unroll
    for (int c = 0; c < 4; ++c) {
      v8bf lo = *(const v8bf*)(qrow + c * 32);
      v8bf hi = *(const v8bf*)(qrow + c * 32 + 16);
      qf[c] = cat16(lo, hi);
    }
  }

  v8f acc[8];
  #pragma unroll
  for (int i = 0; i < 8; ++i) acc[i] = vzero8();
  float m8[8], l8[8];
  #pragma unroll
  for (int v = 0; v < 8; ++v) { m8[v] = -__builtin_inff(); l8[v] = 0.f; }

  const int rowbase = qwb + (half << 3);
  const int kend = q0 + 64;                    // causal upper bound for this tile
  for (int j0 = 0; j0 < kend; j0 += 32) {
    // ---- S = Q K^T for two 16-key subtiles; preload all K-frags, then WMMA
    v8f s0 = vzero8(), s1 = vzero8();
    {
      const __bf16* krow = Kp + (long)(j0 + l15) * DH + half * 8;
      v16bf kf[4];
      #pragma unroll
      for (int c = 0; c < 4; ++c) {
        v8bf lo = *(const v8bf*)(krow + c * 32);
        v8bf hi = *(const v8bf*)(krow + c * 32 + 16);
        kf[c] = cat16(lo, hi);
      }
      #pragma unroll
      for (int c = 0; c < 4; ++c) s0 = wmma_bf16(qf[c], kf[c], s0);
    }
    {
      const __bf16* krow = Kp + (long)(j0 + 16 + l15) * DH + half * 8;
      v16bf kf[4];
      #pragma unroll
      for (int c = 0; c < 4; ++c) {
        v8bf lo = *(const v8bf*)(krow + c * 32);
        v8bf hi = *(const v8bf*)(krow + c * 32 + 16);
        kf[c] = cat16(lo, hi);
      }
      #pragma unroll
      for (int c = 0; c < 4; ++c) s1 = wmma_bf16(qf[c], kf[c], s1);
    }
    // ---- causal mask
    const int jc0 = j0 + l15, jc1 = j0 + 16 + l15;
    #pragma unroll
    for (int v = 0; v < 8; ++v) {
      const int qr = rowbase + v;
      if (jc0 > qr) s0[v] = -__builtin_inff();
      if (jc1 > qr) s1[v] = -__builtin_inff();
    }
    // ---- row max across the 16-lane column group
    float t8[8];
    #pragma unroll
    for (int v = 0; v < 8; ++v) t8[v] = fmaxf(s0[v], s1[v]);
    #pragma unroll
    for (int off = 8; off >= 1; off >>= 1) {
      #pragma unroll
      for (int v = 0; v < 8; ++v) t8[v] = fmaxf(t8[v], __shfl_xor(t8[v], off, 32));
    }
    // ---- online softmax update
    float f8[8];
    #pragma unroll
    for (int v = 0; v < 8; ++v) {
      float mn = fmaxf(m8[v], t8[v]);
      f8[v] = __expf(m8[v] - mn);
      m8[v] = mn;
    }
    float p0[8], p1[8];
    #pragma unroll
    for (int v = 0; v < 8; ++v) {
      p0[v] = __expf(s0[v] - m8[v]);
      p1[v] = __expf(s1[v] - m8[v]);
      l8[v] = l8[v] * f8[v] + p0[v] + p1[v];
    }
    #pragma unroll
    for (int dt = 0; dt < 8; ++dt) {
      #pragma unroll
      for (int v = 0; v < 8; ++v) acc[dt][v] *= f8[v];
    }
    // ---- P: C layout -> LDS -> A layout (per-wave private 1KB)
    #pragma unroll
    for (int v = 0; v < 8; ++v) {
      Pl[wave][(half << 3) + v][l15]      = (__bf16)p0[v];
      Pl[wave][(half << 3) + v][16 + l15] = (__bf16)p1[v];
    }
    __syncthreads();
    v16bf pf;
    {
      v8bf lo = *(const v8bf*)&Pl[wave][l15][half * 8];
      v8bf hi = *(const v8bf*)&Pl[wave][l15][16 + half * 8];
      pf = cat16(lo, hi);
    }
    // ---- O += P @ V (V^T stored [Dh,T] -> contiguous B-frags); preload then WMMA
    const __bf16* vcol = Vp + (long)l15 * NT + j0 + half * 8;
    v16bf vf[8];
    #pragma unroll
    for (int dt = 0; dt < 8; ++dt) {
      const __bf16* vrow = vcol + (long)dt * 16 * NT;
      v8bf lo = *(const v8bf*)(vrow);
      v8bf hi = *(const v8bf*)(vrow + 16);
      vf[dt] = cat16(lo, hi);
    }
    #pragma unroll
    for (int dt = 0; dt < 8; ++dt) acc[dt] = wmma_bf16(pf, vf[dt], acc[dt]);
    __syncthreads();
  }

  // ---- finalize: reduce l across lanes, normalize, store bf16 [B,T,H*Dh]
  #pragma unroll
  for (int off = 8; off >= 1; off >>= 1) {
    #pragma unroll
    for (int v = 0; v < 8; ++v) l8[v] += __shfl_xor(l8[v], off, 32);
  }
  #pragma unroll
  for (int dt = 0; dt < 8; ++dt) {
    const int dh = (dt << 4) + l15;
    #pragma unroll
    for (int v = 0; v < 8; ++v) {
      const long qr = rowbase + v;
      const float o = acc[dt][v] / l8[v];
      attn_out[((long)b * NT + qr) * ND + h * DH + dh] = (__bf16)o;
    }
  }
}

// ---------------- driver ----------------
extern "C" void kernel_launch(void* const* d_in, const int* in_sizes, int n_in,
                              void* d_out, int out_size, void* d_ws, size_t ws_size,
                              hipStream_t stream)
{
  (void)in_sizes; (void)n_in; (void)out_size; (void)ws_size;
  const float* x    = (const float*)d_in[0];
  const float* rope = (const float*)d_in[1];
  const float* anw  = (const float*)d_in[2];
  const float* qkvw = (const float*)d_in[3];
  const float* qnw  = (const float*)d_in[4];
  const float* knw  = (const float*)d_in[5];
  const float* opw  = (const float*)d_in[6];
  const float* mnw  = (const float*)d_in[7];
  const float* miw  = (const float*)d_in[8];
  const float* mow  = (const float*)d_in[9];
  float* out = (float*)d_out;

  // Workspace carve (~201 MB). Region B is time-shared:
  //   qkv_f32 (steps 2-3) -> attn_out_bf16 (steps 4-5) -> mlp_hidden_bf16 (steps 7-8)
  char* ws = (char*)d_ws;
  size_t off = 0;
  auto carve = [&](size_t bytes) -> char* {
    char* p = ws + off; off += (bytes + 255) & ~(size_t)255; return p;
  };
  __bf16* h    = (__bf16*)carve((size_t)NR * ND * 2);        // normed activations (reused)
  char*   regB = carve((size_t)NR * 3 * ND * 4);             // 100.7 MB shared region
  float*  qkv  = (float*)regB;
  __bf16* attn = (__bf16*)regB;
  __bf16* mh   = (__bf16*)regB;
  __bf16* Qb   = (__bf16*)carve((size_t)NR * ND * 2);
  __bf16* Kb   = (__bf16*)carve((size_t)NR * ND * 2);
  __bf16* Vt   = (__bf16*)carve((size_t)NR * ND * 2);
  float*  x2   = (float*)carve((size_t)NR * ND * 4);

  // 1) h = rmsnorm(x) * attn_norm_w  (bf16)
  rmsnorm_to_bf16<<<NR, 256, 0, stream>>>(x, anw, h, ND);
  // 2) qkv = h @ qkv_w^T  (f32)
  gemm_bf16_wmma<<<dim3(3 * ND / 128, NR / 128), 256, 0, stream>>>(
      h, qkvw, (const float*)nullptr, qkv, NR, 3 * ND, ND, 0, 0);
  // 3) RoPE + q/k norm (+scale folded into q), V transpose
  rope_qknorm<<<dim3(NR, NH), 128, 0, stream>>>(qkv, rope, qnw, knw, Qb, Kb, Vt);
  // 4) causal flash attention
  flash_attn_wmma<<<dim3(NT / 64, NB * NH), 128, 0, stream>>>(Qb, Kb, Vt, attn);
  // 5) x2 = x + attn @ out_proj_w^T
  gemm_bf16_wmma<<<dim3(ND / 128, NR / 128), 256, 0, stream>>>(
      attn, opw, x, x2, NR, ND, ND, 0, 0);
  // 6) h = rmsnorm(x2) * mlp_norm_w
  rmsnorm_to_bf16<<<NR, 256, 0, stream>>>(x2, mnw, h, ND);
  // 7) mh = relu(h @ mlp_in_w^T)^2  (bf16)
  gemm_bf16_wmma<<<dim3(NF / 128, NR / 128), 256, 0, stream>>>(
      h, miw, (const float*)nullptr, mh, NR, NF, ND, 1, 1);
  // 8) out = x2 + mh @ mlp_out_w^T  (f32)
  gemm_bf16_wmma<<<dim3(ND / 128, NR / 128), 256, 0, stream>>>(
      mh, mow, x2, out, NR, ND, NF, 0, 0);
}
